// GAT_86852828660381
// MI455X (gfx1250) — matrix operand order repaired and checked by reading, not compiled, not run on hardware
//
#include <hip/hip_runtime.h>
#include <hip/hip_bf16.h>
#include <cstdint>

typedef __attribute__((ext_vector_type(16))) _Float16 v16h;
typedef __attribute__((ext_vector_type(4)))  _Float16 v4h;
typedef __attribute__((ext_vector_type(8)))  float    v8f;

#define N_NODES 100000
#define N_EDGES 1600000

// ---------------------------------------------------------------------------
// CDNA5 async global->LDS copy (VGLOBAL async op, tracked by ASYNCcnt).
// lds_off: byte offset into wave's LDS aperture (low 32 bits of shared ptr).
// ---------------------------------------------------------------------------
__device__ __forceinline__ void async_copy_b128(unsigned lds_off, const void* gptr) {
    asm volatile("global_load_async_to_lds_b128 %0, %1, off"
                 :: "v"(lds_off), "v"(gptr) : "memory");
}
__device__ __forceinline__ void wait_async0() {
    asm volatile("s_wait_asynccnt 0" ::: "memory");
}

// ---------------------------------------------------------------------------
// Utility: fill float buffer with a constant (grid-stride)
// ---------------------------------------------------------------------------
__global__ void fill_f32(float* __restrict__ p, float v, int n) {
    int i = blockIdx.x * blockDim.x + threadIdx.x;
    int stride = gridDim.x * blockDim.x;
    for (; i < n; i += stride) p[i] = v;
}

// ---------------------------------------------------------------------------
// Weight prep.
//   Wp1 / Wp2: B-operand weights pre-permuted into WMMA fragment-major order:
//     Wp[ct][kk][lane][i] = W[kk*32 + (lane<16?0:8) + (i&7) + ((i>>3)<<4)][ct*16 + (lane&15)]
//     (W1 = [w1_src | lin1_w]  64x128 ;  W2 = [w2_src | lin2_w]  64x64)
//   vs/vd: attention projections  vs[i] = dot(w_src[i,:], att_src) etc., so
//   per-node scores are single 64-dot products (saves the x@w_dst GEMM).
// Launch: 1 block x 256 threads (tiny).
// ---------------------------------------------------------------------------
__global__ void prep_weights(
    const float* __restrict__ w1_src, const float* __restrict__ w1_dst,
    const float* __restrict__ att1_src, const float* __restrict__ att1_dst,
    const float* __restrict__ lin1_w,
    const float* __restrict__ w2_src, const float* __restrict__ w2_dst,
    const float* __restrict__ att2_src, const float* __restrict__ att2_dst,
    const float* __restrict__ lin2_w,
    _Float16* __restrict__ Wp1, _Float16* __restrict__ Wp2,
    float* __restrict__ vs1, float* __restrict__ vd1,
    float* __restrict__ vs2, float* __restrict__ vd2)
{
    int t = threadIdx.x;
    for (int idx = t; idx < 8 * 2 * 32 * 16; idx += 256) {      // 64x128
        int i    = idx & 15;
        int lane = (idx >> 4) & 31;
        int kk   = (idx >> 9) & 1;
        int ct   = idx >> 10;
        int col  = ct * 16 + (lane & 15);
        int k    = kk * 32 + ((lane < 16) ? 0 : 8) + (i & 7) + ((i >> 3) << 4);
        float v  = (col < 64) ? w1_src[k * 64 + col] : lin1_w[k * 64 + (col - 64)];
        Wp1[idx] = (_Float16)v;
    }
    for (int idx = t; idx < 4 * 2 * 32 * 16; idx += 256) {      // 64x64
        int i    = idx & 15;
        int lane = (idx >> 4) & 31;
        int kk   = (idx >> 9) & 1;
        int ct   = idx >> 10;
        int col  = ct * 16 + (lane & 15);
        int k    = kk * 32 + ((lane < 16) ? 0 : 8) + (i & 7) + ((i >> 3) << 4);
        float v  = (col < 32) ? w2_src[k * 32 + col] : lin2_w[k * 32 + (col - 32)];
        Wp2[idx] = (_Float16)v;
    }
    if (t < 64) {
        float s = 0.f, d = 0.f;
        for (int j = 0; j < 64; ++j) {
            s += w1_src[t * 64 + j] * att1_src[j];
            d += w1_dst[t * 64 + j] * att1_dst[j];
        }
        vs1[t] = s; vd1[t] = d;
        float s2 = 0.f, d2 = 0.f;
        for (int j = 0; j < 32; ++j) {
            s2 += w2_src[t * 32 + j] * att2_src[j];
            d2 += w2_dst[t * 32 + j] * att2_dst[j];
        }
        vs2[t] = s2; vd2[t] = d2;
    }
}

// ---------------------------------------------------------------------------
// Convert node features f32[N,64] -> fragment-major f16 for the WMMA A operand:
//   Aswz[tile][kk][lane][i],  lane = (row&15) + half*16
//   kr = k&31: half = (kr>>3)&1,  i = (kr&7) | ((kr>=16)<<3)
// Each thread converts 4 consecutive k of one row -> one aligned 8B store.
// ---------------------------------------------------------------------------
__global__ __launch_bounds__(256) void convert_swizzle(
    const float* __restrict__ X, _Float16* __restrict__ Aswz, int n)
{
    int tgl = blockIdx.x * blockDim.x + threadIdx.x;
    if (tgl >= n * 16) return;                 // n*64/4 quads
    int flat  = tgl * 4;
    int row_g = flat >> 6;
    int k0    = flat & 63;
    const float4 v = *(const float4*)(X + (size_t)row_g * 64 + k0);
    int tile = row_g >> 4, row = row_g & 15;
    int kk   = k0 >> 5,    kr  = k0 & 31;
    int half = (kr >> 3) & 1;
    int i0   = (kr & 7) | (((kr >> 4) & 1) << 3);
    int lane = row + half * 16;
    _Float16* dp = Aswz + (size_t)tile * 1024 + (size_t)(kk * 32 + lane) * 16 + i0;
    v4h h4 = { (_Float16)v.x, (_Float16)v.y, (_Float16)v.z, (_Float16)v.w };
    *(v4h*)dp = h4;
}

// ---------------------------------------------------------------------------
// WMMA GEMM: Y[N,C] = X[N,64] @ W[64,C], A/B pre-swizzled to fragment order.
// C is a compile-time constant -> fully unrolled, immediate-offset stores.
// Block = 256 threads = 8 waves, 16 output rows per block.
// Wave 0 stages the 2KB A tile into LDS via async global->LDS (ASYNCcnt),
// then every wave does per-K-half: 1 LDS 32B read + 1 global 32B read + v_wmma.
// ---------------------------------------------------------------------------
template <int C>
__global__ __launch_bounds__(256) void gemm_wmma_f16(
    const _Float16* __restrict__ Aswz, const _Float16* __restrict__ Wp,
    float* __restrict__ Y)
{
    __shared__ __align__(32) _Float16 atile[2 * 32 * 16];   // 2 KB
    const int t = threadIdx.x;
    const int wave = t >> 5, lane = t & 31;

    if (wave == 0) {
        unsigned lbase = ((unsigned)(uintptr_t)&atile[0]) + (unsigned)(lane * 16);
        const char* g = (const char*)(Aswz + (size_t)blockIdx.x * 1024) + lane * 16;
#pragma unroll
        for (int c = 0; c < 4; ++c)                          // 4 * 512B = 2KB
            async_copy_b128(lbase + c * 512u, g + c * 512);
        wait_async0();
    }
    __syncthreads();

    const int m0    = blockIdx.x * 16;
    const int rbase = (lane < 16) ? 0 : 8;
    const int colL  = lane & 15;

#pragma unroll
    for (int ct = wave; ct < (C >> 4); ct += 8) {
        v8f acc = {};
#pragma unroll
        for (int kk = 0; kk < 2; ++kk) {
            v16h a = *(const v16h*)&atile[(kk * 32 + lane) * 16];
            v16h b = *(const v16h*)(Wp + ((size_t)(ct * 2 + kk) * 32 + lane) * 16);
            acc = __builtin_amdgcn_wmma_f32_16x16x32_f16(
                false, a, false, b, (short)0, acc, false, false);
        }
        float* yb = Y + (size_t)(m0 + rbase) * C + (ct << 4) + colL;
#pragma unroll
        for (int r = 0; r < 8; ++r)
            yb[(size_t)r * C] = acc[r];                      // immediate offsets r*C*4
    }
}

// ---------------------------------------------------------------------------
// Per-node attention scalars: a_s[n] = dot(X[n,:64], vs), a_d[n] = dot(X[n,:64], vd)
// One wave32 per node, 2 features/lane, shuffle reduction.
// ---------------------------------------------------------------------------
__global__ __launch_bounds__(256) void node_att(
    const float* __restrict__ X,
    const float* __restrict__ vs, const float* __restrict__ vd,
    float* __restrict__ as_, float* __restrict__ ad_, int n)
{
    int warp = (blockIdx.x * blockDim.x + threadIdx.x) >> 5;
    int lane = threadIdx.x & 31;
    if (warp >= n) return;
    const float* xr = X + (size_t)warp * 64;
    float x0 = xr[lane], x1 = xr[lane + 32];
    float s = x0 * vs[lane] + x1 * vs[lane + 32];
    float d = x0 * vd[lane] + x1 * vd[lane + 32];
#pragma unroll
    for (int off = 16; off > 0; off >>= 1) {
        s += __shfl_xor(s, off, 32);
        d += __shfl_xor(d, off, 32);
    }
    if (lane == 0) { as_[warp] = s; ad_[warp] = d; }
}

// ---------------------------------------------------------------------------
// Edge pass A: e = leaky_relu(a_s[src] + a_d[dst], 0.2); segment max over dst.
// Float atomic max via sign-aware int/uint min/max (m initialized to -inf).
// ---------------------------------------------------------------------------
__global__ void edge_scores(
    const int* __restrict__ src, const int* __restrict__ dst,
    const float* __restrict__ as_, const float* __restrict__ ad_,
    float* __restrict__ e_buf, float* __restrict__ m, int E)
{
    int e = blockIdx.x * blockDim.x + threadIdx.x;
    if (e >= E) return;
    int s = src[e], d = dst[e];
    float v = as_[s] + ad_[d];
    v = (v > 0.f) ? v : 0.2f * v;
    e_buf[e] = v;
    if (v >= 0.f) atomicMax((int*)(m + d), __float_as_int(v));
    else          atomicMin((unsigned int*)(m + d), __float_as_uint(v));
}

// ---------------------------------------------------------------------------
// Edge pass B: ex = exp(e - m[dst]); segment sum over dst. e_buf overwritten.
// ---------------------------------------------------------------------------
__global__ void edge_exp(
    const int* __restrict__ dst, float* __restrict__ e_buf,
    const float* __restrict__ m, float* __restrict__ ssum, int E)
{
    int e = blockIdx.x * blockDim.x + threadIdx.x;
    if (e >= E) return;
    int d = dst[e];
    float ex = expf(e_buf[e] - m[d]);
    e_buf[e] = ex;
    atomicAdd(ssum + d, ex);
}

// ---------------------------------------------------------------------------
// Edge pass C: agg[dst] += (ex / max(ssum[dst],1e-16)) * XS[src].
// One wave32 per edge; compile-time C/LD -> fully unrolled gather + atomics.
// ---------------------------------------------------------------------------
template <int C, int LD>
__global__ __launch_bounds__(256) void edge_aggregate(
    const int* __restrict__ src, const int* __restrict__ dst,
    const float* __restrict__ e_buf, const float* __restrict__ ssum,
    const float* __restrict__ XS, float* __restrict__ agg, int E)
{
    int warp = (blockIdx.x * blockDim.x + threadIdx.x) >> 5;
    int lane = threadIdx.x & 31;
    if (warp >= E) return;
    int s = src[warp], d = dst[warp];
    __builtin_prefetch(XS + (size_t)s * LD + lane, 0, 0);   // global_prefetch_b8
    float alpha = e_buf[warp] / fmaxf(ssum[d], 1e-16f);
#pragma unroll
    for (int ff = 0; ff < C / 32; ++ff) {
        int f = lane + ff * 32;
        atomicAdd(&agg[(size_t)d * C + f], alpha * XS[(size_t)s * LD + f]);
    }
}

// ---------------------------------------------------------------------------
// Layer-1 epilogue (in place): h = relu(agg + b1 + lin1_b + skip), skip = y1[:,64:128]
// ---------------------------------------------------------------------------
__global__ void finish1(
    float* __restrict__ h, const float* __restrict__ b1,
    const float* __restrict__ lin1_b, const float* __restrict__ y1, int n)
{
    int i = blockIdx.x * blockDim.x + threadIdx.x;
    if (i >= n * 64) return;
    int j = i & 63, node = i >> 6;
    float v = h[i] + b1[j] + lin1_b[j] + y1[(size_t)node * 128 + 64 + j];
    h[i] = (v > 0.f) ? v : 0.f;
}

// ---------------------------------------------------------------------------
// Layer-2 epilogue (in place on d_out): out += b2 + lin2_b + skip, skip = y2[:,32:64]
// ---------------------------------------------------------------------------
__global__ void finish2(
    float* __restrict__ out, const float* __restrict__ b2,
    const float* __restrict__ lin2_b, const float* __restrict__ y2, int n)
{
    int i = blockIdx.x * blockDim.x + threadIdx.x;
    if (i >= n * 32) return;
    int o = i & 31, node = i >> 5;
    out[i] += b2[o] + lin2_b[o] + y2[(size_t)node * 64 + 32 + o];
}

// ---------------------------------------------------------------------------
// Host launcher
// ---------------------------------------------------------------------------
extern "C" void kernel_launch(void* const* d_in, const int* in_sizes, int n_in,
                              void* d_out, int out_size, void* d_ws, size_t ws_size,
                              hipStream_t stream) {
    (void)in_sizes; (void)n_in; (void)out_size; (void)ws_size;

    const float* x        = (const float*)d_in[0];
    const int*   ei       = (const int*)d_in[1];
    const float* w1_src   = (const float*)d_in[2];
    const float* w1_dst   = (const float*)d_in[3];
    const float* att1_src = (const float*)d_in[4];
    const float* att1_dst = (const float*)d_in[5];
    const float* b1       = (const float*)d_in[6];
    const float* lin1_w   = (const float*)d_in[7];
    const float* lin1_b   = (const float*)d_in[8];
    const float* w2_src   = (const float*)d_in[9];
    const float* w2_dst   = (const float*)d_in[10];
    const float* att2_src = (const float*)d_in[11];
    const float* att2_dst = (const float*)d_in[12];
    const float* b2       = (const float*)d_in[13];
    const float* lin2_w   = (const float*)d_in[14];
    const float* lin2_b   = (const float*)d_in[15];

    const int* src = ei;
    const int* dst = ei + N_EDGES;
    float* out = (float*)d_out;

    // ---- workspace carve (all 256B aligned) ----
    char* ws = (char*)d_ws;
    size_t off = 0;
    auto carve = [&](size_t bytes) {
        char* p = ws + off;
        off += (bytes + 255) & ~(size_t)255;
        return p;
    };
    float*    y1    = (float*)   carve((size_t)N_NODES * 128 * 4); // [xs|skip]; reused as y2
    float*    h     = (float*)   carve((size_t)N_NODES * 64 * 4);  // agg1 -> h (in place)
    float*    e_buf = (float*)   carve((size_t)N_EDGES * 4);
    _Float16* Aswz  = (_Float16*)carve((size_t)N_NODES * 64 * 2);  // fragment-major A (both layers)
    float*    a_s   = (float*)   carve((size_t)N_NODES * 4);
    float*    a_d   = (float*)   carve((size_t)N_NODES * 4);
    float*    mbuf  = (float*)   carve((size_t)N_NODES * 4);
    float*    ssum  = (float*)   carve((size_t)N_NODES * 4);
    _Float16* Wp1   = (_Float16*)carve((size_t)8192 * 2);
    _Float16* Wp2   = (_Float16*)carve((size_t)4096 * 2);
    float*    vs1   = (float*)   carve(64 * 4);
    float*    vd1   = (float*)   carve(64 * 4);
    float*    vs2   = (float*)   carve(64 * 4);
    float*    vd2   = (float*)   carve(64 * 4);
    float*    y2    = y1;

    const float NEG_INF = -__builtin_huge_valf();
    const int EB  = (N_EDGES + 255) / 256;         // edge-parallel blocks
    const int EWB = N_EDGES * 32 / 256;            // warp-per-edge blocks (exact: 200000)
    const int NWB = N_NODES * 32 / 256;            // warp-per-node blocks (exact: 12500)
    const int MT  = N_NODES / 16;                  // GEMM row tiles (exact: 6250)
    const int CVB = N_NODES * 16 / 256;            // convert blocks (exact: 6250)

    // ---- init ----
    fill_f32<<<2048, 256, 0, stream>>>(h, 0.f, N_NODES * 64);
    fill_f32<<<2048, 256, 0, stream>>>(out, 0.f, N_NODES * 32);
    fill_f32<<<512,  256, 0, stream>>>(mbuf, NEG_INF, N_NODES);
    fill_f32<<<512,  256, 0, stream>>>(ssum, 0.f, N_NODES);
    prep_weights<<<1, 256, 0, stream>>>(w1_src, w1_dst, att1_src, att1_dst, lin1_w,
                                        w2_src, w2_dst, att2_src, att2_dst, lin2_w,
                                        Wp1, Wp2, vs1, vd1, vs2, vd2);

    // ---- layer 1 ----
    convert_swizzle<<<CVB, 256, 0, stream>>>(x, Aswz, N_NODES);
    gemm_wmma_f16<128><<<MT, 256, 0, stream>>>(Aswz, Wp1, y1);
    node_att<<<NWB, 256, 0, stream>>>(x, vs1, vd1, a_s, a_d, N_NODES);
    edge_scores<<<EB, 256, 0, stream>>>(src, dst, a_s, a_d, e_buf, mbuf, N_EDGES);
    edge_exp<<<EB, 256, 0, stream>>>(dst, e_buf, mbuf, ssum, N_EDGES);
    edge_aggregate<64, 128><<<EWB, 256, 0, stream>>>(src, dst, e_buf, ssum,
                                                     y1, h, N_EDGES);
    finish1<<<(N_NODES * 64 + 255) / 256, 256, 0, stream>>>(h, b1, lin1_b, y1, N_NODES);

    // ---- layer 2 ----
    fill_f32<<<512, 256, 0, stream>>>(mbuf, NEG_INF, N_NODES);
    fill_f32<<<512, 256, 0, stream>>>(ssum, 0.f, N_NODES);
    convert_swizzle<<<CVB, 256, 0, stream>>>(h, Aswz, N_NODES);
    gemm_wmma_f16<64><<<MT, 256, 0, stream>>>(Aswz, Wp2, y2);
    node_att<<<NWB, 256, 0, stream>>>(h, vs2, vd2, a_s, a_d, N_NODES);
    edge_scores<<<EB, 256, 0, stream>>>(src, dst, a_s, a_d, e_buf, mbuf, N_EDGES);
    edge_exp<<<EB, 256, 0, stream>>>(dst, e_buf, mbuf, ssum, N_EDGES);
    edge_aggregate<32, 64><<<EWB, 256, 0, stream>>>(src, dst, e_buf, ssum,
                                                    y2, out, N_EDGES);
    finish2<<<(N_NODES * 32 + 255) / 256, 256, 0, stream>>>(out, b2, lin2_b, y2, N_NODES);
}
